// SpatioTemporalPixelEncoder_51805895524591
// MI455X (gfx1250) — matrix operand order repaired and checked by reading, not compiled, not run on hardware
//
#include <hip/hip_runtime.h>
#include <hip/hip_bf16.h>
#include <math.h>

typedef _Float16 v16h __attribute__((ext_vector_type(16)));
typedef _Float16 v8h  __attribute__((ext_vector_type(8)));
typedef float    v8f  __attribute__((ext_vector_type(8)));

#define NB 16
#define NT 2048
#define NF 64
#define NK 128          // [c | s]
#define IMG 224
#define NS 448          // sampled rows/cols (i0 set ++ i1 set)

// ---- workspace layout (bytes) ----
// header: 64 u32 slots (256 B)
//   [0] gmin_x  [1] gmax_x  [2..17] batch_min  [18..33] batch_max  [34] out_min  [35] out_max
#define OFF_APACK  ((size_t)256)
#define APACK_BYTES ((size_t)NB*NT*NK*2)                 // 8 MB
#define OFF_BPACK  (OFF_APACK + APACK_BYTES)
#define OFF_AS     (OFF_BPACK + APACK_BYTES)
#define AS_BYTES   ((size_t)NB*NS*NK*2)                  // 1.75 MB
#define OFF_BS     (OFF_AS + AS_BYTES)
#define OFF_CSUB   (OFF_BS + AS_BYTES)                   // 16*448*448 f32 = 12.8 MB

// ---- ordered-uint encoding: unsigned compare order == float order ----
__device__ __forceinline__ unsigned fkey(float f) {
    unsigned u = __float_as_uint(f);
    return (u & 0x80000000u) ? ~u : (u | 0x80000000u);
}
__device__ __forceinline__ float kdec(unsigned k) {
    return __uint_as_float((k & 0x80000000u) ? (k ^ 0x80000000u) : ~k);
}

// bilinear source coords, align_corners=False (matches reference)
__device__ __forceinline__ void coordf(int o, int& i0, int& i1, float& w) {
    float src = ((float)o + 0.5f) * (2048.0f / 224.0f) - 0.5f;
    src = fminf(fmaxf(src, 0.0f), 2047.0f);
    i0 = (int)floorf(src);
    i1 = min(i0 + 1, 2047);
    w = src - (float)i0;
}

__device__ __forceinline__ v16h load_frag(const _Float16* p0, const _Float16* p1) {
    v8h lo = *(const v8h*)p0;
    v8h hi = *(const v8h*)p1;
    return __builtin_shufflevector(lo, hi, 0,1,2,3,4,5,6,7,8,9,10,11,12,13,14,15);
}

// ---------------- kernel 1: init min/max slots ----------------
__global__ void k_init(unsigned* slots) {
    int i = threadIdx.x;
    if (i < 64) {
        bool isMax = (i == 1) || (i >= 18 && i < 34) || (i == 35);
        slots[i] = isMax ? 0u : 0xFFFFFFFFu;
    }
}

// ---------------- kernel 2: global min/max of x ----------------
__global__ void k_minmax_x(const float* __restrict__ x, int n, unsigned* slots) {
    float mn = 3.4e38f, mx = -3.4e38f;
    for (int i = blockIdx.x * blockDim.x + threadIdx.x; i < n; i += gridDim.x * blockDim.x) {
        float v = x[i];
        mn = fminf(mn, v); mx = fmaxf(mx, v);
    }
    __shared__ float smn[256], smx[256];
    smn[threadIdx.x] = mn; smx[threadIdx.x] = mx;
    __syncthreads();
    for (int s = 128; s > 0; s >>= 1) {
        if (threadIdx.x < s) {
            smn[threadIdx.x] = fminf(smn[threadIdx.x], smn[threadIdx.x + s]);
            smx[threadIdx.x] = fmaxf(smx[threadIdx.x], smx[threadIdx.x + s]);
        }
        __syncthreads();
    }
    if (threadIdx.x == 0) {
        atomicMin(&slots[0], fkey(smn[0]));
        atomicMax(&slots[1], fkey(smx[0]));
    }
}

// ---------------- kernel 3: pack fp16 [c|s] and [c|-s] ----------------
__global__ void k_pack(const float* __restrict__ x, const unsigned* __restrict__ slots,
                       _Float16* __restrict__ Apack, _Float16* __restrict__ Bpack, int n) {
    int idx = blockIdx.x * blockDim.x + threadIdx.x;
    if (idx >= n) return;
    float lo = kdec(slots[0]), hi = kdec(slots[1]);
    float range = hi - lo;
    float v = x[idx];
    float xn = (range < 1e-8f) ? 0.0f : (v - lo) / (range + 1e-8f);
    float c = xn * 2.0f - 1.0f;
    c = fminf(fmaxf(c, -1.0f + 1e-6f), 1.0f - 1e-6f);      // cos(arccos(c)) == c
    float s = sqrtf(fmaxf(0.0f, 1.0f - c * c));            // sin(arccos(c))
    size_t row = (size_t)(idx >> 6);
    int f = idx & 63;
    size_t base = row * NK;
    Apack[base + f]      = (_Float16)c;
    Apack[base + 64 + f] = (_Float16)s;
    Bpack[base + f]      = (_Float16)c;
    Bpack[base + 64 + f] = (_Float16)(-s);
}

// ---------------- kernel 4: fused GEMM (A·Bᵀ) + per-batch min/max ----------------
// One wave -> 16x64 output strip (4 tiles along j, A-fragment reused).
// 65536 waves total, 8 waves/block, 512 blocks per batch (block never spans batches).
__global__ void k_gemm_minmax(const _Float16* __restrict__ Apack,
                              const _Float16* __restrict__ Bpack,
                              unsigned* slots) {
    int wv = threadIdx.x >> 5, lane = threadIdx.x & 31;
    int gw = blockIdx.x * 8 + wv;
    int b   = gw >> 12;            // 4096 waves per batch
    int tl  = gw & 4095;
    int ti  = tl >> 5;             // 0..127
    int tjb = (tl & 31) * 4;       // j-tile base, 0..124
    int m = lane & 15, sel = lane >> 4;

    const _Float16* Ab = Apack + (size_t)b * NT * NK;
    const _Float16* Bb = Bpack + (size_t)b * NT * NK;
    const _Float16* ar  = Ab + (size_t)(ti * 16 + m) * NK;
    const _Float16* br0 = Bb + (size_t)((tjb + 0) * 16 + m) * NK;
    const _Float16* br1 = Bb + (size_t)((tjb + 1) * 16 + m) * NK;
    const _Float16* br2 = Bb + (size_t)((tjb + 2) * 16 + m) * NK;
    const _Float16* br3 = Bb + (size_t)((tjb + 3) * 16 + m) * NK;

    v8f a0 = {}, a1 = {}, a2 = {}, a3 = {};
#pragma unroll
    for (int kb = 0; kb < NK; kb += 32) {
        v16h af  = load_frag(ar  + kb + sel * 16, ar  + kb + sel * 16 + 8);
        v16h bf0 = load_frag(br0 + kb + sel * 16, br0 + kb + sel * 16 + 8);
        v16h bf1 = load_frag(br1 + kb + sel * 16, br1 + kb + sel * 16 + 8);
        v16h bf2 = load_frag(br2 + kb + sel * 16, br2 + kb + sel * 16 + 8);
        v16h bf3 = load_frag(br3 + kb + sel * 16, br3 + kb + sel * 16 + 8);
        a0 = __builtin_amdgcn_wmma_f32_16x16x32_f16(false, af, false, bf0, (short)0, a0, false, false);
        a1 = __builtin_amdgcn_wmma_f32_16x16x32_f16(false, af, false, bf1, (short)0, a1, false, false);
        a2 = __builtin_amdgcn_wmma_f32_16x16x32_f16(false, af, false, bf2, (short)0, a2, false, false);
        a3 = __builtin_amdgcn_wmma_f32_16x16x32_f16(false, af, false, bf3, (short)0, a3, false, false);
    }

    float mn = 3.4e38f, mx = -3.4e38f;
#pragma unroll
    for (int i = 0; i < 8; i++) {
        float lo4 = fminf(fminf(a0[i], a1[i]), fminf(a2[i], a3[i]));
        float hi4 = fmaxf(fmaxf(a0[i], a1[i]), fmaxf(a2[i], a3[i]));
        mn = fminf(mn, lo4); mx = fmaxf(mx, hi4);
    }
    mn *= (1.0f / 64.0f); mx *= (1.0f / 64.0f);   // / F (positive scale preserves order)
#pragma unroll
    for (int off = 16; off >= 1; off >>= 1) {
        mn = fminf(mn, __shfl_xor(mn, off, 32));
        mx = fmaxf(mx, __shfl_xor(mx, off, 32));
    }
    __shared__ float smn[8], smx[8];
    if (lane == 0) { smn[wv] = mn; smx[wv] = mx; }
    __syncthreads();
    if (threadIdx.x == 0) {
        float bmn = smn[0], bmx = smx[0];
#pragma unroll
        for (int i = 1; i < 8; i++) { bmn = fminf(bmn, smn[i]); bmx = fmaxf(bmx, smx[i]); }
        atomicMin(&slots[2 + b],  fkey(bmn));
        atomicMax(&slots[18 + b], fkey(bmx));
    }
}

// ---------------- kernel 5: gather sampled rows for resize cross ----------------
__global__ void k_gather(const _Float16* __restrict__ Apack, const _Float16* __restrict__ Bpack,
                         _Float16* __restrict__ As, _Float16* __restrict__ Bs) {
    int idx = blockIdx.x * blockDim.x + threadIdx.x;   // exactly NB*NS*NK threads
    int k = idx & 127;
    int r = (idx >> 7) % NS;
    int b = idx / (NS * NK);
    int o = (r < IMG) ? r : (r - IMG);
    int i0, i1; float w;
    coordf(o, i0, i1, w);
    int row = (r < IMG) ? i0 : i1;
    size_t src = (size_t)b * NT * NK + (size_t)row * NK + k;
    As[idx] = Apack[src];
    Bs[idx] = Bpack[src];
}

// ---------------- kernel 6: 448x448 sampled GEMM (WMMA) -> Csub ----------------
__global__ void k_gemm_sub(const _Float16* __restrict__ As, const _Float16* __restrict__ Bs,
                           float* __restrict__ Csub) {
    int wv = threadIdx.x >> 5, lane = threadIdx.x & 31;
    int gw = blockIdx.x * 8 + wv;
    if (gw >= NB * 28 * 28) return;                     // uniform per-wave exit
    int b = gw / 784, tl = gw % 784;
    int ti = tl / 28, tj = tl % 28;
    int m = lane & 15, sel = lane >> 4;

    const _Float16* ar = As + (size_t)b * NS * NK + (size_t)(ti * 16 + m) * NK;
    const _Float16* br = Bs + (size_t)b * NS * NK + (size_t)(tj * 16 + m) * NK;
    v8f acc = {};
#pragma unroll
    for (int kb = 0; kb < NK; kb += 32) {
        v16h af = load_frag(ar + kb + sel * 16, ar + kb + sel * 16 + 8);
        v16h bf = load_frag(br + kb + sel * 16, br + kb + sel * 16 + 8);
        acc = __builtin_amdgcn_wmma_f32_16x16x32_f16(false, af, false, bf, (short)0, acc, false, false);
    }
    float* Cb = Csub + (size_t)b * NS * NS;
#pragma unroll
    for (int v = 0; v < 8; v++) {
        int M = v + sel * 8;
        Cb[(size_t)(ti * 16 + M) * NS + (tj * 16 + m)] = acc[v] * (1.0f / 64.0f);
    }
}

// ---------------- kernel 7: per-batch normalize + bilinear resize + global min/max ----
__global__ void k_resize(const float* __restrict__ Csub, const unsigned* __restrict__ slots,
                         float* __restrict__ out, unsigned* oslots, int n) {
    int idx = blockIdx.x * blockDim.x + threadIdx.x;
    float val = 0.0f;
    if (idx < n) {
        int b = idx / (IMG * IMG);
        int rem = idx % (IMG * IMG);
        int oy = rem / IMG, ox = rem % IMG;
        float lo = kdec(slots[2 + b]);
        float hi = kdec(slots[18 + b]);
        float rg = hi - lo;
        const float* Cb = Csub + (size_t)b * NS * NS;
        int i0, i1, j0, j1; float wh, ww;
        coordf(oy, i0, i1, wh);
        coordf(ox, j0, j1, ww);
        float g00 = Cb[(size_t)oy * NS + ox];
        float g01 = Cb[(size_t)oy * NS + IMG + ox];
        float g10 = Cb[(size_t)(IMG + oy) * NS + ox];
        float g11 = Cb[(size_t)(IMG + oy) * NS + IMG + ox];
        if (rg < 1e-8f) {
            g00 = g01 = g10 = g11 = 0.0f;
        } else {
            float inv = 1.0f / (rg + 1e-8f);
            g00 = (g00 - lo) * inv; g01 = (g01 - lo) * inv;
            g10 = (g10 - lo) * inv; g11 = (g11 - lo) * inv;
        }
        float top0 = g00 * (1.0f - wh) + g10 * wh;
        float top1 = g01 * (1.0f - wh) + g11 * wh;
        val = top0 * (1.0f - ww) + top1 * ww;
        out[idx] = val;
    }
    // global min/max of resized image
    __shared__ float smn[256], smx[256];
    smn[threadIdx.x] = (idx < n) ? val : 3.4e38f;
    smx[threadIdx.x] = (idx < n) ? val : -3.4e38f;
    __syncthreads();
    for (int s = 128; s > 0; s >>= 1) {
        if (threadIdx.x < s) {
            smn[threadIdx.x] = fminf(smn[threadIdx.x], smn[threadIdx.x + s]);
            smx[threadIdx.x] = fmaxf(smx[threadIdx.x], smx[threadIdx.x + s]);
        }
        __syncthreads();
    }
    if (threadIdx.x == 0) {
        atomicMin(&oslots[34], fkey(smn[0]));
        atomicMax(&oslots[35], fkey(smx[0]));
    }
}

// ---------------- kernel 8: final global normalize ----------------
__global__ void k_final(float* __restrict__ out, const unsigned* __restrict__ slots, int n) {
    int idx = blockIdx.x * blockDim.x + threadIdx.x;
    if (idx >= n) return;
    float mn = kdec(slots[34]);
    float mx = kdec(slots[35]);
    out[idx] = (out[idx] - mn) / ((mx - mn) + 1e-6f);
}

extern "C" void kernel_launch(void* const* d_in, const int* in_sizes, int n_in,
                              void* d_out, int out_size, void* d_ws, size_t ws_size,
                              hipStream_t stream) {
    const float* x = (const float*)d_in[0];
    float* out = (float*)d_out;
    char* ws = (char*)d_ws;

    unsigned*  slots = (unsigned*)ws;
    _Float16*  Apack = (_Float16*)(ws + OFF_APACK);
    _Float16*  Bpack = (_Float16*)(ws + OFF_BPACK);
    _Float16*  As    = (_Float16*)(ws + OFF_AS);
    _Float16*  Bs    = (_Float16*)(ws + OFF_BS);
    float*     Csub  = (float*)(ws + OFF_CSUB);

    const int nX = NB * NT * NF;       // 2,097,152
    const int nOut = NB * IMG * IMG;   // 802,816

    k_init<<<1, 64, 0, stream>>>(slots);
    k_minmax_x<<<2048, 256, 0, stream>>>(x, nX, slots);
    k_pack<<<(nX + 255) / 256, 256, 0, stream>>>(x, slots, Apack, Bpack, nX);
    // 16 batches * 128 ti * 32 tj-groups = 65536 waves / 8 per block
    k_gemm_minmax<<<8192, 256, 0, stream>>>(Apack, Bpack, slots);
    k_gather<<<(NB * NS * NK) / 256, 256, 0, stream>>>(Apack, Bpack, As, Bs);
    // 16 * 28 * 28 = 12544 waves
    k_gemm_sub<<<(12544 + 7) / 8, 256, 0, stream>>>(As, Bs, Csub);
    k_resize<<<(nOut + 255) / 256, 256, 0, stream>>>(Csub, slots, out, slots, nOut);
    k_final<<<(nOut + 255) / 256, 256, 0, stream>>>(out, slots, nOut);
}
